// ExpertDistillationLoss_17102559773160
// MI455X (gfx1250) — matrix operand order repaired and checked by reading, not compiled, not run on hardware
//
#include <hip/hip_runtime.h>

// ---------------------------------------------------------------------------
// Problem constants (from reference)
// ---------------------------------------------------------------------------
#define H_DIM 2048
#define NTOK  16384       // B*S = 8*2048
#define E_DIM 8
#define R_DIM 16
#define K_MC  3
#define B_DIM 8

typedef __bf16 bf16_t;
typedef __attribute__((ext_vector_type(16))) __bf16        v16bf;
typedef __attribute__((ext_vector_type(8)))  float         v8f;
typedef __attribute__((ext_vector_type(8)))  unsigned int  v8u;

static __device__ __forceinline__ unsigned short f2bf(float f) {
  return __builtin_bit_cast(unsigned short, (bf16_t)f);
}

static __device__ __forceinline__ v8f wmma_bf16(v16bf a, v16bf b, v8f c) {
  // D = A(16x32 bf16) * B(32x16 bf16) + C(16x16 f32)
  return __builtin_amdgcn_wmma_f32_16x16x32_bf16(false, a, false, b,
                                                 (short)0, c, false, false);
}

// ---- Fragment loaders -----------------------------------------------------
// Interleaved swizzled A fragments: phase 0 stored, for each (kb, lane), the
// lane's 16-bf16 teacher fragment at byte (kb*2048 + lane*64) and the student
// fragment at +32.  Each 32B vector load legalizes to 2x ds_load_b128 with
// immediate offsets off one per-lane base pointer.
static __device__ __forceinline__ void
ld_afrag_pair(const char* xc, int lane, int kb, v16bf* at, v16bf* as_) {
  const char* base = xc + kb * 2048 + lane * 64;
  v8u ut = *(const v8u*)(base);
  v8u us = *(const v8u*)(base + 32);
  *at  = __builtin_bit_cast(v16bf, ut);
  *as_ = __builtin_bit_cast(v16bf, us);
}

// Generic (strided) A fragment (16x32 bf16): lane m = lane&15, kh = lane>>4.
// VGPR j<4: K = 2j, 2j+1 (+8*kh) ; j>=4: K = 16+2(j-4), +1 (+8*kh).
static __device__ __forceinline__ v16bf
ld_afrag(const unsigned short* x, int rowstride, int m, int kh, int k0) {
  v8u u;
#pragma unroll
  for (int j = 0; j < 8; ++j) {
    int kk = (j < 4 ? 2 * j : 16 + 2 * (j - 4)) + 8 * kh;
    u[j] = *(const unsigned int*)(x + (size_t)m * rowstride + k0 + kk);
  }
  return __builtin_bit_cast(v16bf, u);
}

// B fragment (32x16 bf16) where B[k][n] = w[n_row][k0+k]; w row-major, stride H.
// VGPR j<4: K = 2j, 2j+1 (+8*kh) ; j>=4: K = 16+2(j-4), +1 (+8*kh).
static __device__ __forceinline__ v16bf
ld_bfrag_rowmajor(const unsigned short* w, int n_row, int kh, int k0) {
  v8u u;
#pragma unroll
  for (int j = 0; j < 8; ++j) {
    int kk = (j < 4 ? 2 * j : 16 + 2 * (j - 4)) + 8 * kh;
    u[j] = *(const unsigned int*)(w + (size_t)n_row * H_DIM + k0 + kk);
  }
  return __builtin_bit_cast(v16bf, u);
}

// B fragment where B[k][n] = buf[(k0+k)*stride + col] (K-major memory).
static __device__ __forceinline__ v16bf
ld_bfrag_kmajor(const unsigned short* buf, int stride, int col, int kh, int k0) {
  v8u u;
#pragma unroll
  for (int j = 0; j < 8; ++j) {
    int kk = (j < 4 ? 2 * j : 16 + 2 * (j - 4)) + 8 * kh;
    unsigned lo = buf[(size_t)(k0 + kk) * stride + col];
    unsigned hi = buf[(size_t)(k0 + kk + 1) * stride + col];
    u[j] = lo | (hi << 16);
  }
  return __builtin_bit_cast(v16bf, u);
}

// ---------------------------------------------------------------------------
// Prep kernels
// ---------------------------------------------------------------------------
__global__ void k_cvt_bf16(const float* __restrict__ s,
                           unsigned short* __restrict__ d, int n) {
  for (int i = blockIdx.x * blockDim.x + threadIdx.x; i < n;
       i += gridDim.x * blockDim.x)
    d[i] = f2bf(s[i]);
}

// Bcat[h][c] : c in [0,128) -> B_s[e][h][r], c in [128,256) -> B_t[e][h][r]
__global__ void k_bcat(const float* __restrict__ Bs, const float* __restrict__ Bt,
                       unsigned short* __restrict__ bcat) {
  int i = blockIdx.x * blockDim.x + threadIdx.x;   // H*256 threads
  int h = i >> 8, c = i & 255;
  float v;
  if (c < 128) { int e = c >> 4, r = c & 15; v = Bs[((size_t)e * H_DIM + h) * R_DIM + r]; }
  else { int c2 = c - 128; int e = c2 >> 4, r = c2 & 15; v = Bt[((size_t)e * H_DIM + h) * R_DIM + r]; }
  bcat[i] = f2bf(v);
}

// Acat[k][c] : c<16 -> A_t[c][k] ; c>=16 -> A_s[c-16][k]
__global__ void k_acat(const float* __restrict__ At, const float* __restrict__ As,
                       unsigned short* __restrict__ acat) {
  int i = blockIdx.x * blockDim.x + threadIdx.x;   // H*32 threads
  int k = i >> 5, c = i & 31;
  float v = (c < 16) ? At[(size_t)c * H_DIM + k] : As[(size_t)(c - 16) * H_DIM + k];
  acat[i] = f2bf(v);
}

__global__ void k_bdiff_zero(const float* __restrict__ bt, const float* __restrict__ bs,
                             float* __restrict__ bdiff, float* __restrict__ acc) {
  int i = blockIdx.x * blockDim.x + threadIdx.x;
  if (i < H_DIM) bdiff[i] = bs[i] - bt[i];
  if (i < 32)    acc[i] = 0.0f;
}

// G[which][e][r][q] : which 0=ss 1=st 2=tt
__global__ void k_gram(const float* __restrict__ Bs, const float* __restrict__ Bt,
                       float* __restrict__ G) {
  int gid = blockIdx.x * blockDim.x + threadIdx.x;   // 3*8*16*16 = 6144
  int which = gid >> 11;
  int rem = gid & 2047;
  int e = rem >> 8, r = (rem >> 4) & 15, q = rem & 15;
  const float* X = (which == 2) ? Bt : Bs;
  const float* Y = (which == 0) ? Bs : Bt;
  float s = 0.0f;
  for (int h = 0; h < H_DIM; ++h)
    s += X[((size_t)e * H_DIM + h) * R_DIM + r] * Y[((size_t)e * H_DIM + h) * R_DIM + q];
  G[gid] = s;
}

// ---------------------------------------------------------------------------
// Main fused kernel: 1024 blocks x 128 threads (4 wave32), 16 tokens/block
// Dynamic LDS layout (byte offsets):
#define SM_XC    0        // interleaved swizzled X_t/X_s fragments (131072 B)
#define SM_PROJ  131072   // 16*256 f32              (16384 B)
#define SM_SUMSQ 147456   // 16 f32
#define SM_AS    147520   // 16*16 f32
#define SM_AT    148544   // 16*16 f32
#define SM_MSE   149568   // 16*8 f32
#define SM_ACCL  150080   // 24 f32
#define SM_DST   150176   // 4 waves * 16*32 bf16    (4096 B)
#define SM_TOTAL 154272
// ---------------------------------------------------------------------------
__global__ __launch_bounds__(128)
void k_main(const float* __restrict__ tg, const float* __restrict__ sg,
            const float* __restrict__ th, const float* __restrict__ sh,
            const int* __restrict__ mask, const float* __restrict__ un,
            const unsigned short* __restrict__ Wt, const unsigned short* __restrict__ Ws,
            const unsigned short* __restrict__ Bcat, const unsigned short* __restrict__ Acat,
            const float* __restrict__ bdiff, const float* __restrict__ G,
            const float* __restrict__ temp_ptr, float* __restrict__ acc_g) {
  extern __shared__ char smem[];
  char*           XC   = smem + SM_XC;
  unsigned int*   XCw  = (unsigned int*)XC;
  float*          proj = (float*)(smem + SM_PROJ);
  float*          sumsq= (float*)(smem + SM_SUMSQ);
  float*          a_s_l= (float*)(smem + SM_AS);
  float*          a_t_l= (float*)(smem + SM_AT);
  float*          mse  = (float*)(smem + SM_MSE);
  float*          accl = (float*)(smem + SM_ACCL);
  unsigned short* dst  = (unsigned short*)(smem + SM_DST);

  const int tid  = threadIdx.x;
  const int wave = tid >> 5;
  const int lane = tid & 31;
  const int ln   = lane & 15;    // A-row / B-col within fragment
  const int kh   = lane >> 4;    // K-half selector
  const size_t row0 = (size_t)blockIdx.x * 16;

  // ---- Phase 0: stage X as bf16, fragment-major, t/s interleaved ----------
  // dword dw -> kb = dw>>9, lane = (dw>>4)&31, which = (dw>>3)&1, j = dw&7.
  // Dword holds the pair X[m][kb*32 + kbase(j,kh)], X[m][..+1];
  // m = lane&15, kh = lane>>4, X = (which ? sh : th).
  for (int dw = tid; dw < 16 * H_DIM; dw += 128) {
    int kb = dw >> 9, rem = dw & 511;
    int lne = rem >> 4, which = (rem >> 3) & 1, j = rem & 7;
    int m = lne & 15, khh = lne >> 4;
    int kbase = (j < 4 ? 2 * j : 16 + 2 * (j - 4)) + 8 * khh;
    size_t src = (row0 + m) * H_DIM + kb * 32 + kbase;
    const float* xp = which ? sh : th;
    XCw[dw] = (unsigned)f2bf(xp[src]) | ((unsigned)f2bf(xp[src + 1]) << 16);
  }
  {
    float* zb = proj;  // proj..accl are contiguous: (SM_DST-SM_PROJ)/4 floats
    for (int i = tid; i < (SM_DST - SM_PROJ) / 4; i += 128) zb[i] = 0.0f;
  }
  __syncthreads();

  // ---- Phase 1: fused GEMMs via WMMA -------------------------------------
  const v8f Z8 = {0.f, 0.f, 0.f, 0.f, 0.f, 0.f, 0.f, 0.f};
  unsigned short* dstw = dst + wave * 512;   // 16x32 bf16 per-wave d-tile
  float sq[8] = {0.f, 0.f, 0.f, 0.f, 0.f, 0.f, 0.f, 0.f};
  v8f accp[16];
#pragma unroll
  for (int i = 0; i < 16; ++i) accp[i] = Z8;

  for (int p = 0; p < 16; ++p) {            // 16 column-pairs per wave
    const int h0 = wave * 512 + p * 32;
#pragma unroll
    for (int sub = 0; sub < 2; ++sub) {
      const int hb = h0 + sub * 16;
      v8f at = Z8, as_ = Z8;
      for (int kb = 0; kb < H_DIM / 32; ++kb) {
        v16bf axt, axs;
        ld_afrag_pair(XC, lane, kb, &axt, &axs);
        v16bf bwt = ld_bfrag_rowmajor(Wt, hb + ln, kh, kb * 32);
        v16bf bws = ld_bfrag_rowmajor(Ws, hb + ln, kh, kb * 32);
        at  = wmma_bf16(axt, bwt, at);
        as_ = wmma_bf16(axs, bws, as_);
      }
      const float bd = bdiff[hb + ln];
#pragma unroll
      for (int v = 0; v < 8; ++v) {
        float d = as_[v] - at[v] + bd;       // C-layout: token = v+8*kh, col = ln
        sq[v] += d * d;
        dstw[(v + 8 * kh) * 32 + sub * 16 + ln] = f2bf(d);
      }
    }
    // proj += d(16x32) * Bcat[h0:h0+32, :]  (LDS is in-order within a wave)
    v16bf ad = ld_afrag(dstw, 32, ln, kh, 0);
#pragma unroll
    for (int ct = 0; ct < 16; ++ct) {
      v16bf bc = ld_bfrag_kmajor(Bcat, 256, ct * 16 + ln, kh, h0);
      accp[ct] = wmma_bf16(ad, bc, accp[ct]);
    }
  }

  // wave 0: a_t = x_t @ A_t^T, a_s = x_s @ A_s^T  (R=16 -> one n-tile each)
  if (wave == 0) {
    v8f aat = Z8, aas = Z8;
    for (int kb = 0; kb < H_DIM / 32; ++kb) {
      v16bf axt, axs;
      ld_afrag_pair(XC, lane, kb, &axt, &axs);
      v16bf bat = ld_bfrag_kmajor(Acat, 32, ln, kh, kb * 32);
      v16bf bas = ld_bfrag_kmajor(Acat, 32, 16 + ln, kh, kb * 32);
      aat = wmma_bf16(axt, bat, aat);
      aas = wmma_bf16(axs, bas, aas);
    }
#pragma unroll
    for (int v = 0; v < 8; ++v) {
      a_t_l[(v + 8 * kh) * 16 + ln] = aat[v];
      a_s_l[(v + 8 * kh) * 16 + ln] = aas[v];
    }
  }

  // flush per-wave partials (ds_add_f32)
#pragma unroll
  for (int v = 0; v < 8; ++v) atomicAdd(&sumsq[v + 8 * kh], sq[v]);
#pragma unroll
  for (int ct = 0; ct < 16; ++ct)
#pragma unroll
    for (int v = 0; v < 8; ++v)
      atomicAdd(&proj[(v + 8 * kh) * 256 + ct * 16 + ln], accp[ct][v]);
  __syncthreads();

  // ---- Phase 2a: mse_all[token][e] --------------------------------------
  {
    const int token = tid >> 3, e = tid & 7;   // 128 threads = 16x8
    float av_s[R_DIM], av_t[R_DIM];
#pragma unroll
    for (int r = 0; r < R_DIM; ++r) {
      av_s[r] = a_s_l[token * 16 + r];
      av_t[r] = a_t_l[token * 16 + r];
    }
    float cross = 0.0f;
#pragma unroll
    for (int r = 0; r < R_DIM; ++r)
      cross += 2.0f * proj[token * 256 + e * 16 + r] * av_s[r]
             - 2.0f * proj[token * 256 + 128 + e * 16 + r] * av_t[r];
    float qss = 0.f, qst = 0.f, qtt = 0.f;
    for (int r = 0; r < R_DIM; ++r)
#pragma unroll
      for (int q = 0; q < R_DIM; ++q) {
        qss += av_s[r] * G[(0 * 8 + e) * 256 + r * 16 + q] * av_s[q];
        qst += av_s[r] * G[(1 * 8 + e) * 256 + r * 16 + q] * av_t[q];
        qtt += av_t[r] * G[(2 * 8 + e) * 256 + r * 16 + q] * av_t[q];
      }
    float quad = 4.0f * qss - 8.0f * qst + 4.0f * qtt;
    mse[token * 8 + e] = sumsq[token] * (1.0f / H_DIM)
                       + (2.0f / H_DIM) * cross + (1.0f / H_DIM) * quad;
  }
  __syncthreads();

  // ---- Phase 2b: per-token MC sampling (method A) + method B --------------
  if (tid < 16) {
    const size_t gt = row0 + tid;
    float p[E_DIM], sgv[E_DIM];
#pragma unroll
    for (int e = 0; e < E_DIM; ++e) { p[e] = tg[gt * E_DIM + e]; sgv[e] = sg[gt * E_DIM + e]; }
    const float mf = (float)mask[gt];
    float feat = 0.f, wsum = 0.f, tc[E_DIM], sc[E_DIM];
#pragma unroll
    for (int e = 0; e < E_DIM; ++e) { tc[e] = 0.f; sc[e] = 0.f; }
    for (int k = 0; k < K_MC; ++k) {
      float best = -1e30f; int bi = 0;
#pragma unroll
      for (int e = 0; e < E_DIM; ++e) {
        float u = un[(size_t)k * (NTOK * E_DIM) + gt * E_DIM + e];
        float gmb = -__logf(-__logf(u * (1.0f - 2e-7f) + 1e-7f));
        float lg = __logf(p[e]) + gmb;
        if (lg > best) { best = lg; bi = e; }
      }
      float w = p[bi];
      feat += mf * w * mse[tid * 8 + bi];
      wsum += mf * w;
      tc[bi] += mf * w;
      sc[bi] += mf * sgv[bi];
      p[bi] *= 0.5f;                              // ALPHA
      float s = 0.f;
#pragma unroll
      for (int e = 0; e < E_DIM; ++e) s += p[e];
      float inv = 1.0f / s;
#pragma unroll
      for (int e = 0; e < E_DIM; ++e) p[e] *= inv;
    }
    atomicAdd(&accl[0], feat);
    atomicAdd(&accl[1], wsum);
#pragma unroll
    for (int e = 0; e < E_DIM; ++e) {
      atomicAdd(&accl[2 + e], tc[e]);
      atomicAdd(&accl[10 + e], sc[e]);
    }
    // Method B
    float tC = fminf(fmaxf(temp_ptr[0], 0.5f), 1.5f);
    float invT = 1.0f / tC;
    float mx = -1e30f;
#pragma unroll
    for (int e = 0; e < E_DIM; ++e) mx = fmaxf(mx, sgv[e] * invT);
    float Z = 0.f;
#pragma unroll
    for (int e = 0; e < E_DIM; ++e) Z += __expf(sgv[e] * invT - mx);
    float lZ = __logf(Z);
    float tkl = 0.f, ent = 0.f;
#pragma unroll
    for (int e = 0; e < E_DIM; ++e) {
      float tgv = tg[gt * E_DIM + e];
      float ls = sgv[e] * invT - mx - lZ;          // log softmax(sg/T)
      tkl += tgv * (__logf(tgv) - ls);
      ent += -sgv[e] * __logf(sgv[e]);
    }
    atomicAdd(&accl[18], tkl);
    atomicAdd(&accl[19], ent);
  }
  __syncthreads();
  if (tid < 20) atomicAdd(&acc_g[tid], accl[tid]);
}

// ---------------------------------------------------------------------------
// Finalize: 7 scalar outputs
// ---------------------------------------------------------------------------
__global__ void k_fin(const float* __restrict__ acc,
                      const float* __restrict__ temp_ptr,
                      float* __restrict__ out) {
  if (threadIdx.x != 0 || blockIdx.x != 0) return;
  float feat_loss = acc[0] / fmaxf(acc[1], 1e-8f);
  float tsum = 0.f, ssum = 0.f;
  for (int e = 0; e < E_DIM; ++e) { tsum += acc[2 + e]; ssum += acc[10 + e]; }
  float ta[E_DIM], sa[E_DIM], ts2 = 0.f, ss2 = 0.f;
  for (int e = 0; e < E_DIM; ++e) {
    ta[e] = acc[2 + e] / tsum + 1e-8f;  ts2 += ta[e];
    sa[e] = acc[10 + e] / ssum + 1e-8f; ss2 += sa[e];
  }
  float cov = 0.f;
  for (int e = 0; e < E_DIM; ++e) {
    float t = ta[e] / ts2, s = sa[e] / ss2;
    cov += t * (__logf(t) - __logf(s));
  }
  cov /= (float)E_DIM;
  float a_total = feat_loss + 0.5f * cov;
  float tC = fminf(fmaxf(temp_ptr[0], 0.5f), 1.5f);
  float tkl = acc[18] / (float)B_DIM;
  float ent_loss = -acc[19] / (float)NTOK;
  float b_total = tkl + 0.1f * ent_loss;
  out[0] = feat_loss; out[1] = cov; out[2] = a_total;
  out[3] = tkl; out[4] = ent_loss; out[5] = b_total; out[6] = tC;
}

// ---------------------------------------------------------------------------
extern "C" void kernel_launch(void* const* d_in, const int* in_sizes, int n_in,
                              void* d_out, int out_size, void* d_ws, size_t ws_size,
                              hipStream_t stream) {
  const float* tg  = (const float*)d_in[0];
  const float* sg  = (const float*)d_in[1];
  const float* th  = (const float*)d_in[2];
  const float* sh  = (const float*)d_in[3];
  const int*   msk = (const int*)d_in[4];
  const float* un  = (const float*)d_in[5];
  const float* W_t = (const float*)d_in[6];
  const float* b_t = (const float*)d_in[7];
  const float* A_t = (const float*)d_in[8];
  const float* B_t = (const float*)d_in[9];
  const float* W_s = (const float*)d_in[10];
  const float* b_s = (const float*)d_in[11];
  const float* A_s = (const float*)d_in[12];
  const float* B_s = (const float*)d_in[13];
  const float* temp= (const float*)d_in[14];

  char* ws = (char*)d_ws;
  float*          acc   = (float*)ws;                                   // 32 f32
  unsigned short* Wt_bf = (unsigned short*)(ws + 256);                  // 8 MB
  unsigned short* Ws_bf = Wt_bf + (size_t)H_DIM * H_DIM;                // 8 MB
  unsigned short* Bcat  = Ws_bf + (size_t)H_DIM * H_DIM;                // 1 MB
  unsigned short* Acat  = Bcat + (size_t)H_DIM * 256;                   // 128 KB
  float*          bdiff = (float*)(Acat + (size_t)H_DIM * 32);          // 8 KB
  float*          G     = bdiff + H_DIM;                                // 24 KB

  k_cvt_bf16<<<2048, 256, 0, stream>>>(W_t, Wt_bf, H_DIM * H_DIM);
  k_cvt_bf16<<<2048, 256, 0, stream>>>(W_s, Ws_bf, H_DIM * H_DIM);
  k_bcat<<<(H_DIM * 256) / 256, 256, 0, stream>>>(B_s, B_t, Bcat);
  k_acat<<<(H_DIM * 32) / 256, 256, 0, stream>>>(A_t, A_s, Acat);
  k_bdiff_zero<<<8, 256, 0, stream>>>(b_t, b_s, bdiff, acc);
  k_gram<<<24, 256, 0, stream>>>(B_s, B_t, G);

  k_main<<<NTOK / 16, 128, SM_TOTAL, stream>>>(tg, sg, th, sh, msk, un,
                                               Wt_bf, Ws_bf, Bcat, Acat,
                                               bdiff, G, temp, acc);
  k_fin<<<1, 32, 0, stream>>>(acc, temp, (float*)d_out);
}
